// RGLRU_74028056314219
// MI455X (gfx1250) — matrix operand order repaired and checked by reading, not compiled, not run on hardware
//
#include <hip/hip_runtime.h>
#include <math.h>

typedef _Float16 f16;
typedef __attribute__((ext_vector_type(16))) _Float16 v16h;
typedef __attribute__((ext_vector_type(8)))  _Float16 v8h;
typedef __attribute__((ext_vector_type(8)))  float    v8f;

#define NB    4                 // batch
#define SEQ   4096
#define DH    1024
#define KIN   1040              // 1024 + 16 pos-emb
#define KPAD  1056              // KIN padded to multiple of 32
#define MTOT  (NB*SEQ)          // 16384
#define SCH   32                // scan chunks
#define SCL   (SEQ/SCH)         // 128 steps per chunk

// ---------------- transpose f32 weight [K][N] -> f16 [N][Kpad] (zero pad) ----
__global__ void k_transpose_f16(const float* __restrict__ W, f16* __restrict__ WT,
                                int K, int N, int Kpad) {
  long long idx = (long long)blockIdx.x * blockDim.x + threadIdx.x;
  long long tot = (long long)N * Kpad;
  long long stride = (long long)gridDim.x * blockDim.x;
  for (; idx < tot; idx += stride) {
    int n = (int)(idx / Kpad), k = (int)(idx % Kpad);
    float v = (k < K) ? W[(long long)k * N + n] : 0.f;
    WT[idx] = (f16)v;
  }
}

// ---------------- build x_cat f16 [MTOT][KPAD]: x | sinusoidal PE | 0 --------
__global__ void k_build_xcat(const float* __restrict__ x, f16* __restrict__ xcat) {
  long long idx = (long long)blockIdx.x * blockDim.x + threadIdx.x;
  long long tot = (long long)MTOT * KPAD;
  long long stride = (long long)gridDim.x * blockDim.x;
  for (; idx < tot; idx += stride) {
    int m = (int)(idx / KPAD), d = (int)(idx % KPAD);
    int s = m & (SEQ - 1);
    float v;
    if (d < 1024) {
      v = x[(long long)m * 1024 + d];
    } else if (d < KIN) {
      int i = d - 1024;
      int pair = i >> 1;
      float div = __expf(-9.210340372f * (float)(2 * pair) / 16.f);
      float ang = (float)s * div;
      v = ((i & 1) ? __cosf(ang) : __sinf(ang)) * 0.1f;
    } else {
      v = 0.f;
    }
    xcat[idx] = (f16)v;
  }
}

__device__ __forceinline__ float sigmoidf(float v) {
  return 1.f / (1.f + __expf(-v));
}

// async global(16B per lane) -> LDS copy, tracked by ASYNCcnt
__device__ __forceinline__ void async_copy_b128(const f16* g, unsigned lds_byte_off) {
  asm volatile("global_load_async_to_lds_b128 %0, %1, off"
               :: "v"(lds_byte_off), "v"(g) : "memory");
}
__device__ __forceinline__ void wait_asynccnt0() {
  asm volatile("s_wait_asynccnt 0x0" ::: "memory");
}

// ---------------- WMMA GEMM: C[M][N] = A[M][K] * BT[N][K]^T + bias -----------
// 128x128 block tile, 8 waves (4m x 2n), each wave 32x64 = 2x4 WMMA tiles.
// A staged via async-to-LDS double buffering (one barrier per K-step).
// MODE 0: write f16 C16 (= x1)
// MODE 1: a = sigmoid(r) * sigmoid(ex0[n])                       (ga path)
// MODE 2: xs = sqrt(1-a^2)*sigmoid(r)*x1, a=ex0[o], x1=ex1[o]    (gx path)
// MODE 3: C32 = r                                                (output GEMM)
template<int MODE>
__global__ void __launch_bounds__(256)
k_gemm_wmma(const f16* __restrict__ A, const f16* __restrict__ BT,
            const float* __restrict__ bias,
            float* __restrict__ C32, f16* __restrict__ C16,
            const float* __restrict__ ex0, const f16* __restrict__ ex1,
            int M, int N, int K)
{
  // two buffers of 128 rows x 32 cols f16, row stride 40 f16 (80B, 16B aligned)
  __shared__ f16 As[2 * 128 * 40];
  const int tid  = threadIdx.x;
  const int lane = tid & 31;
  const int wave = tid >> 5;
  const int wm   = wave & 3;            // m quadrant (32 rows each)
  const int wn   = wave >> 2;           // n half (64 cols each)
  const int bm   = blockIdx.y * 128;
  const int bn   = blockIdx.x * 128;
  const int l16  = lane & 15;
  const int kh   = lane >> 4;           // K-half selector per ISA layout

  v8f acc[2][4];
#pragma unroll
  for (int i = 0; i < 2; ++i)
#pragma unroll
    for (int j = 0; j < 4; ++j) acc[i][j] = {};

  const int ar = tid >> 1;              // staging: 2 threads per row, 32B each
  const int ac = (tid & 1) * 16;        // f16 column offset (0 or 16)
  const unsigned lds_lane = (unsigned)(ar * 80 + ac * 2);   // byte offset in buf
  const long long arow = (long long)(bm + ar) * K + ac;

  // prologue: async-stage first K tile into buffer 0
  {
    const f16* ag = A + arow;
    async_copy_b128(ag,     lds_lane);
    async_copy_b128(ag + 8, lds_lane + 16);
  }

  for (int k0 = 0; k0 < K; k0 += 32) {
    const int cur = (k0 >> 5) & 1;
    wait_asynccnt0();                   // my slice of buf[cur] landed
    __syncthreads();                    // everyone's slice landed, prev reads done

    if (k0 + 32 < K) {                  // prefetch next tile into other buffer
      const f16* ag = A + arow + (k0 + 32);
      const unsigned lo = (unsigned)((cur ^ 1) * 10240) + lds_lane;
      async_copy_b128(ag,     lo);
      async_copy_b128(ag + 8, lo + 16);
    }

    // ---- A fragments from LDS (ISA 16-bit A 16x32 layout) ----
    v16h afrag[2];
#pragma unroll
    for (int tm = 0; tm < 2; ++tm) {
      const f16* ap = &As[cur * 5120 + (wm * 32 + tm * 16 + l16) * 40 + 8 * kh];
      v8h lo = *(const v8h*)(ap);       // K = 8*kh .. +8
      v8h hi = *(const v8h*)(ap + 16);  // K = 16+8*kh .. +8
#pragma unroll
      for (int e = 0; e < 8; ++e) { afrag[tm][e] = lo[e]; afrag[tm][8 + e] = hi[e]; }
    }

    // ---- B fragments direct from global W^T [N][K] (hot in L2) ----
#pragma unroll
    for (int tn = 0; tn < 4; ++tn) {
      const f16* bp = BT + (long long)(bn + wn * 64 + tn * 16 + l16) * K
                         + k0 + 8 * kh;
      v8h lo = *(const v8h*)(bp);
      v8h hi = *(const v8h*)(bp + 16);
      v16h bfrag;
#pragma unroll
      for (int e = 0; e < 8; ++e) { bfrag[e] = lo[e]; bfrag[8 + e] = hi[e]; }
#pragma unroll
      for (int tm = 0; tm < 2; ++tm)
        acc[tm][tn] = __builtin_amdgcn_wmma_f32_16x16x32_f16(
            false, afrag[tm], false, bfrag, (short)0, acc[tm][tn], false, false);
    }
  }

  // ---- fused epilogue on C fragment (m = vgpr + 8*kh, n = lane%16) ----
#pragma unroll
  for (int tm = 0; tm < 2; ++tm) {
#pragma unroll
    for (int tn = 0; tn < 4; ++tn) {
      const int n = bn + wn * 64 + tn * 16 + l16;
      const float bv = bias[n];
#pragma unroll
      for (int v = 0; v < 8; ++v) {
        const int m = bm + wm * 32 + tm * 16 + v + 8 * kh;
        const long long o = (long long)m * N + n;
        float r = acc[tm][tn][v] + bv;
        if (MODE == 0) {
          C16[o] = (f16)r;                                   // x1
        } else if (MODE == 1) {
          C32[o] = sigmoidf(r) * sigmoidf(ex0[n]);           // a
        } else if (MODE == 2) {
          float a  = ex0[o];
          float x1 = (float)ex1[o];
          C32[o] = sqrtf(fmaxf(0.f, 1.f - a * a)) * sigmoidf(r) * x1;  // xs
        } else {
          C32[o] = r;                                        // out
        }
      }
    }
  }
}

// ---------------- chunked parallel linear recurrence -------------------------
// Phase A: per (b,ch,chunk): Aprod = prod a, Xacc = fold of chunk from h=0
__global__ void k_scan_chunk(const float* __restrict__ a, const float* __restrict__ xs,
                             float* __restrict__ Aprod, float* __restrict__ Xacc)
{
  int idx = blockIdx.x * blockDim.x + threadIdx.x;   // (b*SCH + c)*DH + ch
  if (idx >= NB * DH * SCH) return;
  int ch  = idx % DH;
  int rem = idx / DH;
  int c   = rem % SCH;
  int b   = rem / SCH;
  long long p = ((long long)b * SEQ + (long long)c * SCL) * DH + ch;
  float A = 1.f, X = 0.f;
  for (int s = 0; s < SCL; ++s, p += DH) {
    float av = a[p];
    A *= av;
    X = av * X + xs[p];
  }
  Aprod[idx] = A;
  Xacc[idx]  = X;
}

// Phase B: per (b,ch): combine chunk summaries, emit chunk start states + h_last
__global__ void k_scan_combine(const float* __restrict__ Aprod, const float* __restrict__ Xacc,
                               const float* __restrict__ h_prev,
                               float* __restrict__ hstart, float* __restrict__ h_last)
{
  int idx = blockIdx.x * blockDim.x + threadIdx.x;   // b*DH + ch
  if (idx >= NB * DH) return;
  int b = idx / DH, ch = idx % DH;
  float h = h_prev[idx];
  for (int c = 0; c < SCH; ++c) {
    long long q = ((long long)b * SCH + c) * DH + ch;
    hstart[q] = h;
    h = Aprod[q] * h + Xacc[q];
  }
  h_last[idx] = h;
}

// Phase C: per (b,ch,chunk): replay chunk from its start state, write h (f16)
__global__ void k_scan_emit(const float* __restrict__ a, const float* __restrict__ xs,
                            const float* __restrict__ hstart, f16* __restrict__ h16)
{
  int idx = blockIdx.x * blockDim.x + threadIdx.x;
  if (idx >= NB * DH * SCH) return;
  int ch  = idx % DH;
  int rem = idx / DH;
  int c   = rem % SCH;
  int b   = rem / SCH;
  long long p = ((long long)b * SEQ + (long long)c * SCL) * DH + ch;
  float h = hstart[idx];
  for (int s = 0; s < SCL; ++s, p += DH) {
    h = a[p] * h + xs[p];
    h16[p] = (f16)h;
  }
}

extern "C" void kernel_launch(void* const* d_in, const int* in_sizes, int n_in,
                              void* d_out, int out_size, void* d_ws, size_t ws_size,
                              hipStream_t stream) {
  const float* x       = (const float*)d_in[0];
  const float* h_prev  = (const float*)d_in[1];
  const float* a_logit = (const float*)d_in[2];
  const float* W_in    = (const float*)d_in[3];
  const float* b_in    = (const float*)d_in[4];
  const float* W_gx    = (const float*)d_in[5];
  const float* b_gx    = (const float*)d_in[6];
  const float* W_ga    = (const float*)d_in[7];
  const float* b_ga    = (const float*)d_in[8];
  const float* W_out   = (const float*)d_in[9];
  const float* b_out   = (const float*)d_in[10];
  float* out = (float*)d_out;                 // [MTOT*DH] out, then [NB*DH] h_last

  // ---- workspace carve-out (256B aligned) ----
  char* w = (char*)d_ws;
  auto carve = [&](size_t bytes) -> char* {
    char* p = w;
    w += (bytes + 255) & ~(size_t)255;
    return p;
  };
  f16*   wInT   = (f16*)  carve((size_t)DH * KPAD * sizeof(f16));
  f16*   wGxT   = (f16*)  carve((size_t)DH * DH   * sizeof(f16));
  f16*   wGaT   = (f16*)  carve((size_t)DH * DH   * sizeof(f16));
  f16*   wOutT  = (f16*)  carve((size_t)DH * DH   * sizeof(f16));
  f16*   xcat   = (f16*)  carve((size_t)MTOT * KPAD * sizeof(f16));
  f16*   x1h    = (f16*)  carve((size_t)MTOT * DH * sizeof(f16));
  float* abuf   = (float*)carve((size_t)MTOT * DH * sizeof(float));
  float* xsbuf  = (float*)carve((size_t)MTOT * DH * sizeof(float));
  f16*   h16    = (f16*)  carve((size_t)MTOT * DH * sizeof(f16));
  float* Aprod  = (float*)carve((size_t)NB * DH * SCH * sizeof(float));
  float* Xacc   = (float*)carve((size_t)NB * DH * SCH * sizeof(float));
  float* hstart = (float*)carve((size_t)NB * DH * SCH * sizeof(float));

  // 1) weights -> f16 transposed (zero-padded K for W_in)
  {
    int blk = 256;
    k_transpose_f16<<<(DH * KPAD + blk - 1) / blk, blk, 0, stream>>>(W_in,  wInT,  KIN, DH, KPAD);
    k_transpose_f16<<<(DH * DH   + blk - 1) / blk, blk, 0, stream>>>(W_gx,  wGxT,  DH,  DH, DH);
    k_transpose_f16<<<(DH * DH   + blk - 1) / blk, blk, 0, stream>>>(W_ga,  wGaT,  DH,  DH, DH);
    k_transpose_f16<<<(DH * DH   + blk - 1) / blk, blk, 0, stream>>>(W_out, wOutT, DH,  DH, DH);
  }

  // 2) x_cat = [x | PE | 0] in f16
  {
    long long tot = (long long)MTOT * KPAD;
    k_build_xcat<<<(unsigned)((tot + 255) / 256), 256, 0, stream>>>(x, xcat);
  }

  dim3 grid(DH / 128, MTOT / 128);     // (8, 128)
  dim3 block(256);

  // 3) x1 = x_cat @ W_in + b_in                (f16 out)
  k_gemm_wmma<0><<<grid, block, 0, stream>>>(xcat, wInT, b_in,
                                             nullptr, x1h, nullptr, nullptr,
                                             MTOT, DH, KPAD);
  // 4) a = sigmoid(x1 @ W_ga + b_ga) * sigmoid(a_logit)
  k_gemm_wmma<1><<<grid, block, 0, stream>>>(x1h, wGaT, b_ga,
                                             abuf, nullptr, a_logit, nullptr,
                                             MTOT, DH, DH);
  // 5) xs = sqrt(1-a^2) * sigmoid(x1 @ W_gx + b_gx) * x1
  k_gemm_wmma<2><<<grid, block, 0, stream>>>(x1h, wGxT, b_gx,
                                             xsbuf, nullptr, abuf, x1h,
                                             MTOT, DH, DH);
  // 6) chunked recurrence -> h (f16), h_last -> tail of d_out
  k_scan_chunk  <<<(NB * DH * SCH) / 256, 256, 0, stream>>>(abuf, xsbuf, Aprod, Xacc);
  k_scan_combine<<<(NB * DH) / 256,       256, 0, stream>>>(Aprod, Xacc, h_prev,
                                                            hstart, out + (size_t)MTOT * DH);
  k_scan_emit   <<<(NB * DH * SCH) / 256, 256, 0, stream>>>(abuf, xsbuf, hstart, h16);
  // 7) out = h @ W_out + b_out
  k_gemm_wmma<3><<<grid, block, 0, stream>>>(h16, wOutT, b_out,
                                             out, nullptr, nullptr, nullptr,
                                             MTOT, DH, DH);
}
